// ResNet4D_38972533244237
// MI455X (gfx1250) — matrix operand order repaired and checked by reading, not compiled, not run on hardware
//
#include <hip/hip_runtime.h>

typedef _Float16 h16;
typedef __attribute__((ext_vector_type(16))) _Float16 v16h;
typedef __attribute__((ext_vector_type(8)))  _Float16 v8h;
typedef __attribute__((ext_vector_type(8)))  float    v8f;

// ---------------------------------------------------------------------------
// Weight repack: f32 [co][ci][tap]  ->  f16 [co][tap][ci], rows padded to
// Kpad = roundup32(Ci*KV) with zeros. Makes the WMMA A-side K dimension
// (tap-major, ci-minor) contiguous so A fragments are two aligned 16B loads.
// ---------------------------------------------------------------------------
__global__ void repack_w_k(const float* __restrict__ src, h16* __restrict__ dst,
                           int Co, int Ci, int KV, int Kpad)
{
    long total = (long)Co * Kpad;
    long idx   = (long)blockIdx.x * blockDim.x + threadIdx.x;
    if (idx >= total) return;
    int rowk = (int)(idx % Kpad);
    int co   = (int)(idx / Kpad);
    h16 v = (h16)0.f;
    if (rowk < Ci * KV) {
        int ci  = rowk % Ci;       // dst row is tap*Ci + ci
        int tap = rowk / Ci;
        v = (h16)src[((long)co * Ci + ci) * KV + tap];
    }
    dst[idx] = v;
}

// ---------------------------------------------------------------------------
// Implicit-GEMM 4D convolution via V_WMMA_F32_16X16X32_F16 (wave32).
// One wave per block; tile = 64 out-channels (4 WMMA M-tiles) x 16 positions.
//
// Activations are POSITION-MAJOR f16: act[p][c] (c fastest). With the K
// ordering tap-major/channel-minor, a B fragment (one position, 16 consecutive
// channels per lane) is two aligned 16B loads; the epilogue writes one aligned
// v8h (8 consecutive channels) per accumulator tile per lane.
//
//   KS==7 (conv1, Ci=1): K == taps, per-element base-7 decode (compile-time
//                        divisors), chunks of 32 taps, f32 spatial input.
//   KS==3/1 (Ci % 32 == 0): four compile-time tap loops with hoisted bounds
//                        check + base address, inner chunks of 32 channels.
//
// Fragment layouts used here (C/D as in CDNA5 ISA; A/B K order chosen by this code):
//   A 16x32 f16: lane l -> row m=l&15, 16 consecutive K slots (+16 lanes>=16), this code's order
//   B 32x16 f16: lane l -> col n=l&15, 16 consecutive K slots (+16 lanes>=16), this code's order
//   C/D 16x16 f32: VGPR r, lane l -> row (l<16 ? r : r+8), col l&15
// ---------------------------------------------------------------------------
template<typename TIn, int KS, int STRIDE, int PAD, bool RELU, bool ADDSKIP>
__global__ __launch_bounds__(32)
void conv4d_wmma(const TIn* __restrict__ in, const h16* __restrict__ wt,
                 const h16* __restrict__ skip, h16* __restrict__ out,
                 int Ci, int Co,
                 int iT, int iD, int iH, int iW,
                 int oT, int oD, int oH, int oW)
{
    constexpr int KV = KS * KS * KS * KS;

    const int  lane   = threadIdx.x;
    const int  Ktot   = Ci * KV;
    const int  Kpad   = (Ktot + 31) & ~31;
    const long P      = (long)oT * oD * oH * oW;
    const int  coBase = blockIdx.y * 64;
    const long pBase  = (long)blockIdx.x * 16;

    const int  n      = lane & 15;
    const long p      = pBase + n;
    const bool pvalid = (p < P);

    // Decompose this lane's output position once.
    int pp = pvalid ? (int)p : 0;
    int ow = pp % oW; int t1 = pp / oW;
    int oh = t1 % oH; int t2 = t1 / oH;
    int od = t2 % oD; int ot = t2 / oD;
    const int iwB = ow * STRIDE - PAD;
    const int ihB = oh * STRIDE - PAD;
    const int idB = od * STRIDE - PAD;
    const int itB = ot * STRIDE - PAD;

    const int koff  = (lane & 16) ? 16 : 0;   // B-side slot base
    const int kaoff = (lane & 16) ? 16 : 0;   // A-side slot base

    const v8f zero = {0.f,0.f,0.f,0.f,0.f,0.f,0.f,0.f};
    v8f acc[4];
    #pragma unroll
    for (int t = 0; t < 4; ++t) acc[t] = zero;

    // Per-tile weight row base pointers (hoisted).
    const h16* wrow[4];
    #pragma unroll
    for (int t = 0; t < 4; ++t)
        wrow[t] = wt + (long)(coBase + t * 16 + n) * Kpad + kaoff;

    // One 32-slot K chunk: load all 4 A fragments (clauseable), then 4 WMMAs.
    auto do_chunk = [&](const v16h& bfrag, int wofs) {
        v16h afr[4];
        #pragma unroll
        for (int t = 0; t < 4; ++t) {
            const h16* rowp = wrow[t] + wofs;
            v8h lo = *(const v8h*)(rowp);
            v8h hi = *(const v8h*)(rowp + 8);
            afr[t] = __builtin_shufflevector(lo, hi,
                        0,1,2,3,4,5,6,7,8,9,10,11,12,13,14,15);
        }
        #pragma unroll
        for (int t = 0; t < 4; ++t)
            acc[t] = __builtin_amdgcn_wmma_f32_16x16x32_f16(
                         false, afr[t], false, bfrag, (short)0, acc[t], false, false);
    };

    if constexpr (KS == 7) {
        // conv1: Ci == 1, K == taps; f32 spatial input, base-7 decode.
        for (int kb = 0; kb < Kpad; kb += 32) {
            v16h b;
            #pragma unroll
            for (int j = 0; j < 16; ++j) {
                int k = kb + koff + j;
                float v = 0.f;
                if (pvalid && k < Ktot) {
                    int r  = k;
                    int dt = r / 343; r -= dt * 343;
                    int dd = r / 49;  r -= dd * 49;
                    int dh = r / 7;   int dw = r - dh * 7;
                    int it = itB + dt, id2 = idB + dd;
                    int ih = ihB + dh, iw  = iwB + dw;
                    if ((unsigned)it < (unsigned)iT && (unsigned)id2 < (unsigned)iD &&
                        (unsigned)ih < (unsigned)iH && (unsigned)iw  < (unsigned)iW) {
                        v = (float)in[(((long)it * iD + id2) * iH + ih) * iW + iw];
                    }
                }
                b[j] = (h16)v;
            }
            do_chunk(b, kb);
        }
    } else {
        // Layer convs: tap loops (compile-time bounds), channel chunks inner.
        // Input is position-major [p][c]; one B fragment = two 16B loads.
        for (int dt = 0; dt < KS; ++dt) {
            int it = itB + dt;  bool bt = (unsigned)it  < (unsigned)iT;
        for (int dd = 0; dd < KS; ++dd) {
            int id2 = idB + dd; bool bd = (unsigned)id2 < (unsigned)iD;
        for (int dh = 0; dh < KS; ++dh) {
            int ih = ihB + dh;  bool bh = (unsigned)ih  < (unsigned)iH;
        for (int dw = 0; dw < KS; ++dw) {
            int iw = iwB + dw;  bool bw = (unsigned)iw  < (unsigned)iW;
            const bool inb = pvalid && bt && bd && bh && bw;
            if (__ballot(inb) == 0ULL) continue;      // wave-uniform skip
            const long sbase  = (((long)it * iD + id2) * iH + ih) * iW + iw;
            const int  tapIdx = ((dt * KS + dd) * KS + dh) * KS + dw;
            const h16* prow   = (const h16*)in + sbase * (long)Ci + koff;
            for (int cb = 0; cb < Ci; cb += 32) {
                v16h b;
                #pragma unroll
                for (int j = 0; j < 16; ++j) b[j] = (h16)0.f;
                if (inb) {
                    v8h lo = *(const v8h*)(prow + cb);
                    v8h hi = *(const v8h*)(prow + cb + 8);
                    b = __builtin_shufflevector(lo, hi,
                            0,1,2,3,4,5,6,7,8,9,10,11,12,13,14,15);
                }
                do_chunk(b, tapIdx * Ci + cb);
            }
        }}}}
    }

    // ---- Epilogue: residual add, ReLU, vectorized f16 store --------------
    if (pvalid) {
        const long rowbase = p * (long)Co;
        #pragma unroll
        for (int t = 0; t < 4; ++t) {
            int co0 = coBase + t * 16 + ((lane & 16) ? 8 : 0);
            v8h sv;
            if (ADDSKIP) sv = *(const v8h*)(skip + rowbase + co0);
            v8h res;
            #pragma unroll
            for (int r = 0; r < 8; ++r) {
                float v = acc[t][r];
                if (ADDSKIP) v += (float)sv[r];
                if (RELU)    v  = v > 0.f ? v : 0.f;
                res[r] = (h16)v;
            }
            *(v8h*)(out + rowbase + co0) = res;
        }
    }
}

// ---------------------------------------------------------------------------
// 4D max-pool, kernel 3, stride 2, pad 1; position-major [p][c] f16 layout
// (adjacent threads = adjacent channels -> fully coalesced).
// ---------------------------------------------------------------------------
__global__ void maxpool4d_k(const h16* __restrict__ in, h16* __restrict__ out,
                            int C, int iT, int iD, int iH, int iW,
                            int oT, int oD, int oH, int oW)
{
    long total = (long)C * oT * oD * oH * oW;
    long idx   = (long)blockIdx.x * blockDim.x + threadIdx.x;
    if (idx >= total) return;
    int  c = (int)(idx % C);
    long t = idx / C;
    int w  = t % oW; t /= oW;
    int h  = t % oH; t /= oH;
    int d  = t % oD; t /= oD;
    int tt = (int)t;

    float m = -3.4e38f;
    for (int a = 0; a < 3; ++a) { int it = tt*2-1+a; if ((unsigned)it >= (unsigned)iT) continue;
    for (int b = 0; b < 3; ++b) { int id2 = d*2-1+b; if ((unsigned)id2 >= (unsigned)iD) continue;
    for (int e = 0; e < 3; ++e) { int ih = h*2-1+e; if ((unsigned)ih >= (unsigned)iH) continue;
    for (int f = 0; f < 3; ++f) { int iw = w*2-1+f; if ((unsigned)iw >= (unsigned)iW) continue;
        long pin = (((long)it * iD + id2) * iH + ih) * iW + iw;
        float v = (float)in[pin * C + c];
        m = v > m ? v : m;
    }}}}
    out[idx] = (h16)m;
}

// ---------------------------------------------------------------------------
// Per-channel spatial mean over position-major [p][c]: one block per channel.
// ---------------------------------------------------------------------------
__global__ void channel_mean_k(const h16* __restrict__ in, float* __restrict__ means,
                               int C, int P)
{
    __shared__ float red[256];
    int c = blockIdx.x;
    float s = 0.f;
    for (int i = threadIdx.x; i < P; i += blockDim.x)
        s += (float)in[(long)i * C + c];
    red[threadIdx.x] = s;
    __syncthreads();
    for (int w = blockDim.x >> 1; w > 0; w >>= 1) {
        if (threadIdx.x < w) red[threadIdx.x] += red[threadIdx.x + w];
        __syncthreads();
    }
    if (threadIdx.x == 0) means[c] = red[0] / (float)P;
}

// ---------------------------------------------------------------------------
// FC: out[0] = dot(means, fc_w) + fc_b   (512 -> 1)
// ---------------------------------------------------------------------------
__global__ void fc_k(const float* __restrict__ means, const float* __restrict__ w,
                     const float* __restrict__ b, float* __restrict__ out)
{
    __shared__ float red[512];
    int t = threadIdx.x;
    red[t] = means[t] * w[t];
    __syncthreads();
    for (int s = 256; s > 0; s >>= 1) {
        if (t < s) red[t] += red[t + s];
        __syncthreads();
    }
    if (t == 0) out[0] = red[0] + b[0];
}

// ---------------------------------------------------------------------------
// Host side
// ---------------------------------------------------------------------------
template<typename TIn, int KS, int S, int PD, bool RELU, bool SK>
static inline void conv_stage(hipStream_t st, const TIn* in, const float* w,
                              h16* wscr, const h16* skip, h16* out,
                              int Ci, int Co,
                              int iT, int iD, int iH, int iW,
                              int oT, int oD, int oH, int oW)
{
    constexpr int KV = KS * KS * KS * KS;
    int  Ktot = Ci * KV;
    int  Kpad = (Ktot + 31) & ~31;
    long wtot = (long)Co * Kpad;
    repack_w_k<<<dim3((unsigned)((wtot + 255) / 256)), dim3(256), 0, st>>>(
        w, wscr, Co, Ci, KV, Kpad);
    long P = (long)oT * oD * oH * oW;
    dim3 g((unsigned)((P + 15) / 16), (unsigned)(Co / 64), 1);
    conv4d_wmma<TIn, KS, S, PD, RELU, SK><<<g, dim3(32, 1, 1), 0, st>>>(
        in, wscr, skip, out, Ci, Co, iT, iD, iH, iW, oT, oD, oH, oW);
}

extern "C" void kernel_launch(void* const* d_in, const int* in_sizes, int n_in,
                              void* d_out, int out_size, void* d_ws, size_t ws_size,
                              hipStream_t stream)
{
    (void)in_sizes; (void)n_in; (void)out_size; (void)ws_size;

    const float* x       = (const float*)d_in[0];
    const float* conv1_w = (const float*)d_in[1];
    const float* l1_w1   = (const float*)d_in[2];
    const float* l1_w2   = (const float*)d_in[3];
    const float* l2_w1   = (const float*)d_in[4];
    const float* l2_w2   = (const float*)d_in[5];
    const float* l2_wd   = (const float*)d_in[6];
    const float* l3_w1   = (const float*)d_in[7];
    const float* l3_w2   = (const float*)d_in[8];
    const float* l3_wd   = (const float*)d_in[9];
    const float* l4_w1   = (const float*)d_in[10];
    const float* l4_w2   = (const float*)d_in[11];
    const float* l4_wd   = (const float*)d_in[12];
    const float* fc_w    = (const float*)d_in[13];
    const float* fc_b    = (const float*)d_in[14];
    float* out = (float*)d_out;

    // Stage shapes
    const int T0=49, D0=49, H0=57, W0=49;                                   // input
    const int T1=25, D1=25, H1=29, W1=25; const long P1=(long)T1*D1*H1*W1;  // conv1
    const int T2=13, D2=13, H2=15, W2=13; const long P2=(long)T2*D2*H2*W2;  // pool / layer1
    const int T3=7,  D3_=7, H3=8,  W3=7;  const long P3=(long)T3*D3_*H3*W3; // layers 2-4

    // Workspace arena, 256B-aligned sub-buffers. Activations are [p][c] f16.
    char*  wsb = (char*)d_ws;
    size_t off = 0;
    auto alloc = [&](size_t bytes) -> char* {
        char* pt = wsb + off;
        off += (bytes + 255) & ~(size_t)255;
        return pt;
    };
    // Repacked-weight scratch (max: l4_w2 = 512 * 41472 halves = 42.5 MB),
    // reused across convs (stream-serialized).
    h16* wscr  = (h16*)alloc((size_t)512 * 41472 * 2);
    h16* c1out = (h16*)alloc((size_t)64  * P1 * 2);
    h16* pout  = (h16*)alloc((size_t)64  * P2 * 2);
    h16* l1y   = (h16*)alloc((size_t)64  * P2 * 2);
    h16* l1o   = (h16*)alloc((size_t)64  * P2 * 2);
    h16* l2y   = (h16*)alloc((size_t)128 * P3 * 2);
    h16* l2s   = (h16*)alloc((size_t)128 * P3 * 2);
    h16* l2o   = (h16*)alloc((size_t)128 * P3 * 2);
    h16* l3y   = (h16*)alloc((size_t)256 * P3 * 2);
    h16* l3s   = (h16*)alloc((size_t)256 * P3 * 2);
    h16* l3o   = (h16*)alloc((size_t)256 * P3 * 2);
    h16* l4y   = (h16*)alloc((size_t)512 * P3 * 2);
    h16* l4s   = (h16*)alloc((size_t)512 * P3 * 2);
    h16* l4o   = (h16*)alloc((size_t)512 * P3 * 2);
    float* means = (float*)alloc(512 * sizeof(float));

    // conv1: 1->64, 7^4, s2, p3, ReLU (f32 spatial input read directly)
    conv_stage<float, 7, 2, 3, true, false>(stream, x, conv1_w, wscr, nullptr, c1out,
                                            1, 64, T0, D0, H0, W0, T1, D1, H1, W1);

    // maxpool 3^4, s2, p1
    {
        long total = 64 * P2;
        maxpool4d_k<<<dim3((unsigned)((total + 255) / 256)), dim3(256), 0, stream>>>(
            c1out, pout, 64, T1, D1, H1, W1, T2, D2, H2, W2);
    }

    // layer1: 64->64, stride 1, identity skip
    conv_stage<h16, 3, 1, 1, true,  false>(stream, pout, l1_w1, wscr, nullptr, l1y,
                                           64, 64, T2, D2, H2, W2, T2, D2, H2, W2);
    conv_stage<h16, 3, 1, 1, true,  true >(stream, l1y, l1_w2, wscr, pout, l1o,
                                           64, 64, T2, D2, H2, W2, T2, D2, H2, W2);

    // layer2: 64->128, stride 2, 1x1 downsample
    conv_stage<h16, 1, 2, 0, false, false>(stream, l1o, l2_wd, wscr, nullptr, l2s,
                                           64, 128, T2, D2, H2, W2, T3, D3_, H3, W3);
    conv_stage<h16, 3, 2, 1, true,  false>(stream, l1o, l2_w1, wscr, nullptr, l2y,
                                           64, 128, T2, D2, H2, W2, T3, D3_, H3, W3);
    conv_stage<h16, 3, 1, 1, true,  true >(stream, l2y, l2_w2, wscr, l2s, l2o,
                                           128, 128, T3, D3_, H3, W3, T3, D3_, H3, W3);

    // layer3: 128->256, stride 1, 1x1 downsample
    conv_stage<h16, 1, 1, 0, false, false>(stream, l2o, l3_wd, wscr, nullptr, l3s,
                                           128, 256, T3, D3_, H3, W3, T3, D3_, H3, W3);
    conv_stage<h16, 3, 1, 1, true,  false>(stream, l2o, l3_w1, wscr, nullptr, l3y,
                                           128, 256, T3, D3_, H3, W3, T3, D3_, H3, W3);
    conv_stage<h16, 3, 1, 1, true,  true >(stream, l3y, l3_w2, wscr, l3s, l3o,
                                           256, 256, T3, D3_, H3, W3, T3, D3_, H3, W3);

    // layer4: 256->512, stride 1, 1x1 downsample
    conv_stage<h16, 1, 1, 0, false, false>(stream, l3o, l4_wd, wscr, nullptr, l4s,
                                           256, 512, T3, D3_, H3, W3, T3, D3_, H3, W3);
    conv_stage<h16, 3, 1, 1, true,  false>(stream, l3o, l4_w1, wscr, nullptr, l4y,
                                           256, 512, T3, D3_, H3, W3, T3, D3_, H3, W3);
    conv_stage<h16, 3, 1, 1, true,  true >(stream, l4y, l4_w2, wscr, l4s, l4o,
                                           512, 512, T3, D3_, H3, W3, T3, D3_, H3, W3);

    // global mean + FC
    channel_mean_k<<<dim3(512), dim3(256), 0, stream>>>(l4o, means, 512, (int)P3);
    fc_k<<<dim3(1), dim3(512), 0, stream>>>(means, fc_w, fc_b, out);
}
